// BasicVQVAE_80367428043243
// MI455X (gfx1250) — matrix operand order repaired and checked by reading, not compiled, not run on hardware
//
#include <hip/hip_runtime.h>
#include <stdint.h>

#if __has_include(<hip/amd_detail/amd_gfx1250_TDM.h>)
#define TDM_6ARG 1
#else
#define TDM_6ARG 0
#endif

// ---------------------------------------------------------------------------
// Types for CDNA5 WMMA (wave32): bf16 A/B fragments (8 VGPRs), f32 C/D (8 VGPRs)
// ---------------------------------------------------------------------------
typedef __attribute__((ext_vector_type(16))) __bf16        v16bf;
typedef __attribute__((ext_vector_type(8)))  float         v8f;
typedef __attribute__((ext_vector_type(4)))  unsigned int  u32x4;
typedef __attribute__((ext_vector_type(8)))  int           i32x8;
typedef __attribute__((ext_vector_type(4)))  int           i32x4;

union FragU { v16bf v; u32x4 q[2]; };
union TdmG0 { u32x4 v; unsigned int d[4]; };
union TdmG1 { i32x8 v; unsigned long long q[4]; };

__device__ __forceinline__ unsigned short f32_to_bf16_bits(float f) {
    unsigned u = __float_as_uint(f);
    unsigned r = u + 0x7FFFu + ((u >> 16) & 1u);   // round-to-nearest-even
    return (unsigned short)(r >> 16);
}

__device__ __forceinline__ unsigned float_sortable(float f) {
    unsigned u = __float_as_uint(f);
    return (u & 0x80000000u) ? ~u : (u | 0x80000000u);
}

// ---------------------------------------------------------------------------
// TDM: async 2D bf16 tile load global -> LDS (Tensor Data Mover, TENSORcnt).
// Tile is tile_d1 rows x tile_d0 elements; row stride in memory = stride elems.
// LDS destination is padded: 16 DWORDs (64B = 32 bf16) then 4 DWORDs pad
// -> LDS row pitch 40 halfs, matching the As/Bs tile layout below.
// ---------------------------------------------------------------------------
__device__ __forceinline__ void tdm_load_2d_bf16(
    unsigned lds_off, const void* gptr,
    unsigned tensor_d0, unsigned tensor_d1,
    unsigned long long stride_elems,
    unsigned tile_d0, unsigned tile_d1)
{
    unsigned long long ga = (unsigned long long)(uintptr_t)gptr;
    TdmG0 g0; TdmG1 g1;
    g0.d[0] = 1u;                                   // count=1 (valid user D#)
    g0.d[1] = lds_off;                              // lds_addr (bytes)
    g0.d[2] = (unsigned)(ga & 0xFFFFFFFFull);       // global_addr[31:0]
    g0.d[3] = (unsigned)((ga >> 32) & 0x01FFFFFFull)// global_addr[56:32]
            | (2u << 30);                           // type=2 ("image")
    g1.q[0] = (1ull << 16)                          // data_size: 2 bytes
            | (1ull << 20)                          // pad_enable
            | (3ull << 22)                          // pad_interval: 16 DWORDs
            | (3ull << 25)                          // pad_amount: 4 DWORDs
            | ((unsigned long long)(tensor_d0 & 0xFFFFu) << 48);
    g1.q[1] = ((unsigned long long)(tensor_d0 >> 16) & 0xFFFFull)
            | ((unsigned long long)tensor_d1 << 16)
            | ((unsigned long long)(tile_d0 & 0xFFFFu) << 48);
    g1.q[2] = ((unsigned long long)(tile_d1 & 0xFFFFu))
            | ((stride_elems & 0xFFFFFFFFull) << 32);
    g1.q[3] = (stride_elems >> 32) & 0xFFFFull;     // dim1_stride = 0 (2D)
    i32x4 z4 = {};
#if TDM_6ARG
    i32x8 z8 = {};
    __builtin_amdgcn_tensor_load_to_lds(g0.v, g1.v, z4, z4, z8, 0);
#else
    __builtin_amdgcn_tensor_load_to_lds(g0.v, g1.v, z4, z4, 0);
#endif
}

// ---------------------------------------------------------------------------
// f32 -> bf16-bits elementwise conversion (grid-stride)
// ---------------------------------------------------------------------------
__global__ __launch_bounds__(256) void k_cvt_bf16(const float* __restrict__ in,
                                                  unsigned short* __restrict__ out,
                                                  long long n) {
    long long i = (long long)blockIdx.x * 256 + threadIdx.x;
    long long stride = (long long)gridDim.x * 256;
    for (; i < n; i += stride) out[i] = f32_to_bf16_bits(in[i]);
}

// ---------------------------------------------------------------------------
// f32 [K][N] -> bf16-bits [N][K] transpose+convert, 32x32 tiles via LDS.
// Makes all GEMM B operands [N][K] so the hot loops use 128-bit LDS traffic
// and TDM-loadable tiles.
// ---------------------------------------------------------------------------
__global__ __launch_bounds__(256) void k_cvt_t(const float* __restrict__ in,
                                               unsigned short* __restrict__ out,
                                               int K, int N) {
    __shared__ float t[32][33];
    const int n0 = blockIdx.x * 32, k0 = blockIdx.y * 32;
    const int c = threadIdx.x & 31, r4 = threadIdx.x >> 5;
    #pragma unroll
    for (int p = 0; p < 4; ++p) {
        int r = r4 + p * 8;
        t[r][c] = in[(long long)(k0 + r) * N + n0 + c];
    }
    __syncthreads();
    #pragma unroll
    for (int p = 0; p < 4; ++p) {
        int r = r4 + p * 8;
        out[(long long)(n0 + r) * K + k0 + c] = f32_to_bf16_bits(t[c][r]);
    }
}

// ---------------------------------------------------------------------------
// Tiled bf16 WMMA GEMM:  C[M,N] = act(A[M,K] @ Bt^T + bias),  Bt is [N,K].
// Block: 128x128 tile, Kstep=32, 256 threads = 8 waves; wave -> 32x64 subtile.
// Tiles staged by TDM (wave 0 issues, double-buffered, s_wait_tensorcnt).
// Requires M%128==0, N%128==0, K%32==0 (true for all shapes here).
// ---------------------------------------------------------------------------
template<int RELU, int OUT_F32>
__global__ __launch_bounds__(256) void k_gemm_bf16(
    const unsigned short* __restrict__ A,    // [M,K]
    const unsigned short* __restrict__ Bt,   // [N,K]
    const float* __restrict__ bias,          // [N]
    void* __restrict__ Cout,                 // f32 or bf16 [M,N]
    int M, int N, int K)
{
    __shared__ unsigned short As[2][128][40];   // [m][k], 8-half pad
    __shared__ unsigned short Bs[2][128][40];   // [n][k]

    const int tid  = threadIdx.x;
    const int lane = tid & 31;
    const int wave = tid >> 5;
    const int wm   = wave & 3;     // 4 wave-rows x 32 rows
    const int wn   = wave >> 2;    // 2 wave-cols x 64 cols
    const int blockM = blockIdx.y * 128;
    const int blockN = blockIdx.x * 128;
    const bool issuer = (wave == 0);

    v8f vzero = {};
    v8f acc[2][4];
    #pragma unroll
    for (int i = 0; i < 2; ++i)
        #pragma unroll
        for (int j = 0; j < 4; ++j) acc[i][j] = vzero;

    auto issue = [&](int buf, int k0) {
        if (issuer) {
            tdm_load_2d_bf16((unsigned)(uintptr_t)&As[buf][0][0],
                             A + (size_t)blockM * K + k0,
                             (unsigned)K, 128u, (unsigned long long)K, 32u, 128u);
            tdm_load_2d_bf16((unsigned)(uintptr_t)&Bs[buf][0][0],
                             Bt + (size_t)blockN * K + k0,
                             (unsigned)K, 128u, (unsigned long long)K, 32u, 128u);
        }
    };

    auto compute = [&](int buf) {
        FragU a[2], b[4];
        const int am = wm * 32 + (lane & 15);
        const int ak = (lane >= 16) ? 8 : 0;       // lanes16-31: K=8..15 / 24..31
        #pragma unroll
        for (int ti = 0; ti < 2; ++ti) {
            a[ti].q[0] = *(const u32x4*)&As[buf][am + ti * 16][ak];
            a[ti].q[1] = *(const u32x4*)&As[buf][am + ti * 16][16 + ak];
        }
        const int bn = wn * 64 + (lane & 15);
        const int bk = (lane >= 16) ? 8 : 0;       // lanes16-31: K=8..15 / 24..31
        #pragma unroll
        for (int tj = 0; tj < 4; ++tj) {
            b[tj].q[0] = *(const u32x4*)&Bs[buf][bn + tj * 16][bk];
            b[tj].q[1] = *(const u32x4*)&Bs[buf][bn + tj * 16][16 + bk];
        }
        #pragma unroll
        for (int ti = 0; ti < 2; ++ti)
            #pragma unroll
            for (int tj = 0; tj < 4; ++tj)
                acc[ti][tj] = __builtin_amdgcn_wmma_f32_16x16x32_bf16(
                    false, a[ti].v, false, b[tj].v,
                    (short)0, acc[ti][tj], false, false);
    };

    const int nsteps = K / 32;
    issue(0, 0);
    int cur = 0;
    for (int s = 0; s < nsteps; ++s) {
        if (s + 1 < nsteps) {
            issue(cur ^ 1, (s + 1) * 32);                 // prefetch next tile
            if (issuer) __builtin_amdgcn_s_wait_tensorcnt(2);
        } else {
            if (issuer) __builtin_amdgcn_s_wait_tensorcnt(0);
        }
        __syncthreads();
        compute(cur);
        __syncthreads();
        cur ^= 1;
    }

    // ---- epilogue: bias + optional ReLU, write f32 or bf16
    #pragma unroll
    for (int ti = 0; ti < 2; ++ti) {
        int row_base = blockM + wm * 32 + ti * 16 + ((lane >= 16) ? 8 : 0);
        #pragma unroll
        for (int tj = 0; tj < 4; ++tj) {
            int col = blockN + wn * 64 + tj * 16 + (lane & 15);
            float bv = bias[col];
            #pragma unroll
            for (int r = 0; r < 8; ++r) {
                float v = acc[ti][tj][r] + bv;
                if (RELU) v = v > 0.0f ? v : 0.0f;
                long long off = (long long)(row_base + r) * N + col;
                if (OUT_F32) ((float*)Cout)[off] = v;
                else         ((unsigned short*)Cout)[off] = f32_to_bf16_bits(v);
            }
        }
    }
}

// ---------------------------------------------------------------------------
// Per-code squared L2 norm of codebook rows (f32)
// ---------------------------------------------------------------------------
__global__ __launch_bounds__(256) void k_code_norms(const float* __restrict__ cb,
                                                    float* __restrict__ norms, int E) {
    __shared__ float red[256];
    const float* row = cb + (long long)blockIdx.x * E;
    float s = 0.0f;
    for (int e = threadIdx.x; e < E; e += 256) { float v = row[e]; s += v * v; }
    red[threadIdx.x] = s; __syncthreads();
    for (int w = 128; w > 0; w >>= 1) {
        if (threadIdx.x < w) red[threadIdx.x] += red[threadIdx.x + w];
        __syncthreads();
    }
    if (threadIdx.x == 0) norms[blockIdx.x] = red[0];
}

// ---------------------------------------------------------------------------
// VQ argmin: per 128 z_e rows, sweep all codes via bf16 WMMA (TDM staged),
//   dist = ||c||^2 - 2 * (z_e . c)   (||z_e||^2 row-constant -> dropped)
// running min as sortable (float<<32)|code key; ds_min_u64 reduction in LDS.
// ---------------------------------------------------------------------------
__global__ __launch_bounds__(256) void k_vq_argmin(
    const unsigned short* __restrict__ Ze,   // [8192, K] bf16 bits
    const unsigned short* __restrict__ Cb,   // [NCODES, K] bf16 bits
    const float* __restrict__ norms,         // [NCODES]
    unsigned int* __restrict__ idx,          // [8192]
    int NCODES, int K)
{
    __shared__ unsigned short As[2][128][40];
    __shared__ unsigned short Bs[2][128][40];
    __shared__ unsigned long long best[128];

    const int tid  = threadIdx.x;
    const int lane = tid & 31;
    const int wave = tid >> 5;
    const int wm   = wave & 3;
    const int wn   = wave >> 2;
    const int blockM = blockIdx.x * 128;
    const bool issuer = (wave == 0);

    if (tid < 128) best[tid] = 0xFFFFFFFFFFFFFFFFull;

    unsigned long long mk[2][8];
    #pragma unroll
    for (int ti = 0; ti < 2; ++ti)
        #pragma unroll
        for (int r = 0; r < 8; ++r) mk[ti][r] = 0xFFFFFFFFFFFFFFFFull;

    v8f vzero = {};
    v8f acc[2][4];

    const int KS = K / 32, NB = NCODES / 128, nsteps = NB * KS;

    auto issue = [&](int buf, int step) {
        if (issuer) {
            int nb = step / KS, ks = step % KS;
            tdm_load_2d_bf16((unsigned)(uintptr_t)&As[buf][0][0],
                             Ze + (size_t)blockM * K + ks * 32,
                             (unsigned)K, 128u, (unsigned long long)K, 32u, 128u);
            tdm_load_2d_bf16((unsigned)(uintptr_t)&Bs[buf][0][0],
                             Cb + (size_t)nb * 128 * K + ks * 32,
                             (unsigned)K, 128u, (unsigned long long)K, 32u, 128u);
        }
    };

    auto compute = [&](int buf) {
        FragU a[2], b[4];
        const int am = wm * 32 + (lane & 15);
        const int ak = (lane >= 16) ? 8 : 0;
        #pragma unroll
        for (int ti = 0; ti < 2; ++ti) {
            a[ti].q[0] = *(const u32x4*)&As[buf][am + ti * 16][ak];
            a[ti].q[1] = *(const u32x4*)&As[buf][am + ti * 16][16 + ak];
        }
        const int bn = wn * 64 + (lane & 15);
        const int bk = (lane >= 16) ? 8 : 0;
        #pragma unroll
        for (int tj = 0; tj < 4; ++tj) {
            b[tj].q[0] = *(const u32x4*)&Bs[buf][bn + tj * 16][bk];
            b[tj].q[1] = *(const u32x4*)&Bs[buf][bn + tj * 16][16 + bk];
        }
        #pragma unroll
        for (int ti = 0; ti < 2; ++ti)
            #pragma unroll
            for (int tj = 0; tj < 4; ++tj)
                acc[ti][tj] = __builtin_amdgcn_wmma_f32_16x16x32_bf16(
                    false, a[ti].v, false, b[tj].v,
                    (short)0, acc[ti][tj], false, false);
    };

    issue(0, 0);
    int cur = 0;
    for (int s = 0; s < nsteps; ++s) {
        const int ks = s % KS;
        if (ks == 0) {
            #pragma unroll
            for (int i = 0; i < 2; ++i)
                #pragma unroll
                for (int j = 0; j < 4; ++j) acc[i][j] = vzero;
        }
        if (s + 1 < nsteps) {
            issue(cur ^ 1, s + 1);
            if (issuer) __builtin_amdgcn_s_wait_tensorcnt(2);
        } else {
            if (issuer) __builtin_amdgcn_s_wait_tensorcnt(0);
        }
        __syncthreads();
        compute(cur);
        if (ks == KS - 1) {
            const int blockN = (s / KS) * 128;
            #pragma unroll
            for (int tj = 0; tj < 4; ++tj) {
                int col = blockN + wn * 64 + tj * 16 + (lane & 15);
                float nrm = norms[col];
                #pragma unroll
                for (int ti = 0; ti < 2; ++ti) {
                    #pragma unroll
                    for (int r = 0; r < 8; ++r) {
                        float dist = nrm - 2.0f * acc[ti][tj][r];
                        unsigned long long key =
                            ((unsigned long long)float_sortable(dist) << 32) | (unsigned)col;
                        if (key < mk[ti][r]) mk[ti][r] = key;
                    }
                }
            }
        }
        __syncthreads();
        cur ^= 1;
    }

    // cross-wave reduction via LDS 64-bit atomic min
    #pragma unroll
    for (int ti = 0; ti < 2; ++ti) {
        int rl = wm * 32 + ti * 16 + ((lane >= 16) ? 8 : 0);
        #pragma unroll
        for (int r = 0; r < 8; ++r)
            atomicMin(&best[rl + r], mk[ti][r]);
    }
    __syncthreads();
    if (tid < 128)
        idx[blockM + tid] = (unsigned)(best[tid] & 0xFFFFFFFFu);
}

// ---------------------------------------------------------------------------
// z_q gather (bf16): zq[row] = codebook[idx[row]]
// ---------------------------------------------------------------------------
__global__ __launch_bounds__(256) void k_gather(const unsigned short* __restrict__ cb,
                                                const unsigned int* __restrict__ idx,
                                                unsigned short* __restrict__ zq, int E) {
    int row = blockIdx.x;
    unsigned code = idx[row];
    const unsigned* src = (const unsigned*)(cb + (long long)code * E);
    unsigned*       dst = (unsigned*)(zq + (long long)row * E);
    for (int t = threadIdx.x; t < E / 2; t += 256) dst[t] = src[t];
}

// ---------------------------------------------------------------------------
// per-row sum((z_q - z_e)^2) in f32 + code-usage histogram
// ---------------------------------------------------------------------------
__global__ __launch_bounds__(256) void k_vq_stats(const float* __restrict__ cb_f32,
                                                  const float* __restrict__ ze_f32,
                                                  const unsigned int* __restrict__ idx,
                                                  float* __restrict__ counts,
                                                  float* __restrict__ accum, int E) {
    __shared__ float red[256];
    int row = blockIdx.x;
    unsigned code = idx[row];
    const float* c = cb_f32 + (long long)code * E;
    const float* z = ze_f32 + (long long)row * E;
    float s = 0.0f;
    for (int e = threadIdx.x; e < E; e += 256) { float d = c[e] - z[e]; s += d * d; }
    red[threadIdx.x] = s; __syncthreads();
    for (int w = 128; w > 0; w >>= 1) {
        if (threadIdx.x < w) red[threadIdx.x] += red[threadIdx.x + w];
        __syncthreads();
    }
    if (threadIdx.x == 0) {
        atomicAdd(accum, red[0]);
        atomicAdd(&counts[code], 1.0f);
    }
}

__global__ __launch_bounds__(256) void k_zero(float* __restrict__ p, int n) {
    int i = blockIdx.x * 256 + threadIdx.x;
    if (i < n) p[i] = 0.0f;
}

// vq_loss = (1 + COMMIT) * mean((z_q - z_e)^2); perplexity = exp(-sum p log(p+eps))
__global__ __launch_bounds__(256) void k_finalize(const float* __restrict__ counts,
                                                  const float* __restrict__ accum,
                                                  float* __restrict__ out_tail,
                                                  int ncodes, float inv_batch, float inv_elems) {
    __shared__ float red[256];
    float s = 0.0f;
    for (int k = threadIdx.x; k < ncodes; k += 256) {
        float p = counts[k] * inv_batch;
        s += p * __logf(p + 1e-10f);
    }
    red[threadIdx.x] = s; __syncthreads();
    for (int w = 128; w > 0; w >>= 1) {
        if (threadIdx.x < w) red[threadIdx.x] += red[threadIdx.x + w];
        __syncthreads();
    }
    if (threadIdx.x == 0) {
        float msq = accum[0] * inv_elems;
        out_tail[0] = 1.25f * msq;        // q_latent + 0.25*e_latent (equal in fwd)
        out_tail[1] = __expf(-red[0]);
    }
}

// ---------------------------------------------------------------------------
// Host launcher
// ---------------------------------------------------------------------------
extern "C" void kernel_launch(void* const* d_in, const int* in_sizes, int n_in,
                              void* d_out, int out_size, void* d_ws, size_t ws_size,
                              hipStream_t stream) {
    (void)in_sizes; (void)n_in; (void)out_size; (void)ws_size;

    const int B = 8192, X = 4096, H = 4096, Z = 512, E = 512, NC = 8192;

    const float* x      = (const float*)d_in[0];
    const float* enc_w1 = (const float*)d_in[1];
    const float* enc_b1 = (const float*)d_in[2];
    const float* enc_w2 = (const float*)d_in[3];
    const float* enc_b2 = (const float*)d_in[4];
    const float* pre_w  = (const float*)d_in[5];
    const float* pre_b  = (const float*)d_in[6];
    const float* cbf    = (const float*)d_in[7];
    const float* dec_w1 = (const float*)d_in[8];
    const float* dec_b1 = (const float*)d_in[9];
    const float* dec_w2 = (const float*)d_in[10];
    const float* dec_b2 = (const float*)d_in[11];

    // workspace carve-out (256B aligned)
    char* base = (char*)d_ws;
    size_t off = 0;
    auto alloc = [&](size_t bytes) -> void* {
        off = (off + 255) & ~(size_t)255;
        void* p = base + off;
        off += bytes;
        return p;
    };
    unsigned short* xb   = (unsigned short*)alloc((size_t)B * X * 2);   // x bf16 [B][X]
    unsigned short* wbig = (unsigned short*)alloc((size_t)X * H * 2);   // enc_w1^T, reused: dec_w2^T
    unsigned short* hb   = (unsigned short*)alloc((size_t)B * H * 2);   // h, reused: hd
    unsigned short* wsm  = (unsigned short*)alloc((size_t)H * Z * 2);   // enc_w2^T, reused: dec_w1^T
    unsigned short* zb   = (unsigned short*)alloc((size_t)B * Z * 2);
    unsigned short* pwb  = (unsigned short*)alloc((size_t)Z * E * 2);   // pre_w^T
    float*          zef  = (float*)alloc((size_t)B * E * 4);            // z_e f32
    unsigned short* zeb  = (unsigned short*)alloc((size_t)B * E * 2);
    unsigned short* cbb  = (unsigned short*)alloc((size_t)NC * E * 2);
    unsigned short* zqb  = (unsigned short*)alloc((size_t)B * E * 2);
    float*          norms  = (float*)alloc((size_t)NC * 4);
    unsigned int*   idx    = (unsigned int*)alloc((size_t)B * 4);
    float*          counts = (float*)alloc((size_t)(NC + 1) * 4);
    float*          accum  = counts + NC;

    const dim3 blk(256);
    const int CVTG = 4096;

    // encoder
    k_cvt_bf16<<<CVTG, blk, 0, stream>>>(x, xb, (long long)B * X);
    k_cvt_t<<<dim3(H/32, X/32), blk, 0, stream>>>(enc_w1, wbig, X, H);   // -> [H][X]
    k_gemm_bf16<1,0><<<dim3(H/128, B/128), blk, 0, stream>>>(xb, wbig, enc_b1, hb, B, H, X);
    k_cvt_t<<<dim3(Z/32, H/32), blk, 0, stream>>>(enc_w2, wsm, H, Z);    // -> [Z][H]
    k_gemm_bf16<0,0><<<dim3(Z/128, B/128), blk, 0, stream>>>(hb, wsm, enc_b2, zb, B, Z, H);
    k_cvt_t<<<dim3(E/32, Z/32), blk, 0, stream>>>(pre_w, pwb, Z, E);     // -> [E][Z]
    k_gemm_bf16<0,1><<<dim3(E/128, B/128), blk, 0, stream>>>(zb, pwb, pre_b, zef, B, E, Z);
    k_cvt_bf16<<<CVTG, blk, 0, stream>>>(zef, zeb, (long long)B * E);

    // vector quantization (codebook is already [NC][E] = [N][K])
    k_cvt_bf16<<<CVTG, blk, 0, stream>>>(cbf, cbb, (long long)NC * E);
    k_code_norms<<<NC, blk, 0, stream>>>(cbf, norms, E);
    k_vq_argmin<<<B / 128, blk, 0, stream>>>(zeb, cbb, norms, idx, NC, E);
    k_gather<<<B, blk, 0, stream>>>(cbb, idx, zqb, E);
    k_zero<<<(NC + 1 + 255) / 256, blk, 0, stream>>>(counts, NC + 1);
    k_vq_stats<<<B, blk, 0, stream>>>(cbf, zef, idx, counts, accum, E);

    // decoder (straight-through forward value == z_q)
    k_cvt_t<<<dim3(H/32, E/32), blk, 0, stream>>>(dec_w1, wsm, E, H);    // -> [H][E]
    k_gemm_bf16<1,0><<<dim3(H/128, B/128), blk, 0, stream>>>(zqb, wsm, dec_b1, hb, B, H, E);
    k_cvt_t<<<dim3(X/32, H/32), blk, 0, stream>>>(dec_w2, wbig, H, X);   // -> [X][H]
    k_gemm_bf16<0,1><<<dim3(X/128, B/128), blk, 0, stream>>>(hb, wbig, dec_b2, d_out, B, X, H);

    // scalars
    float* out_tail = (float*)d_out + (long long)B * X;
    k_finalize<<<1, blk, 0, stream>>>(counts, accum, out_tail, NC,
                                      1.0f / (float)B, 1.0f / ((float)B * (float)E));
}
